// AttentionBlock_66168266162626
// MI455X (gfx1250) — matrix operand order repaired and checked by reading, not compiled, not run on hardware
//
#include <hip/hip_runtime.h>
#include <hip/hip_bf16.h>

// ---------------------------------------------------------------------------
// AttentionBlock for MI455X (gfx1250, wave32, WMMA bf16).
// Pipeline: GroupNorm -> QKV gemms -> 2-pass flash attention -> proj+residual.
// All matmuls: v_wmma_f32_16x16x32_bf16 (bf16 in, f32 accumulate).
// QKV B-tiles are staged to LDS with the Tensor Data Mover
// (tensor_load_to_lds + s_wait_tensorcnt) when the builtin is available.
// Workspace layout (bytes), total ~162.3 MB (assumes ws_size is sufficient):
//   h    [B,C,N] bf16 : 32M      q_t [B,N,C] bf16 : 32M
//   k    [B,C,N] bf16 : 32M      v_t [B,N,C] bf16 : 32M
//   o_t  [B,N,C] bf16 : 32M      wq/wk/wv/wp bf16 : 2M
//   mean/rstd/m/l f32 : ~260K
// ---------------------------------------------------------------------------

typedef __attribute__((ext_vector_type(16))) __bf16 v16bf;
typedef __attribute__((ext_vector_type(8)))  float  v8f;
typedef __attribute__((ext_vector_type(4)))  unsigned int v4u;
typedef __attribute__((ext_vector_type(4)))  int    v4i;
typedef __attribute__((ext_vector_type(8)))  int    v8i;

#define B_  8
#define C_  512
#define N_  4096
#define G_  32
#define CPG 16
#define EPS_ 1e-6f
#define SCALE_ 0.044194173824159216f   // 512^-0.5

#if __has_builtin(__builtin_amdgcn_tensor_load_to_lds) && \
    __has_builtin(__builtin_amdgcn_s_wait_tensorcnt)
#define HAVE_TDM 1
#else
#define HAVE_TDM 0
#endif

__device__ __forceinline__ __bf16 f2bf(float f) { return (__bf16)f; }

__device__ __forceinline__ v8f zero8() {
  v8f z;
#pragma unroll
  for (int i = 0; i < 8; ++i) z[i] = 0.f;
  return z;
}

__device__ __forceinline__ v8f wmma_bf16(v16bf a, v16bf b, v8f c) {
  return __builtin_amdgcn_wmma_f32_16x16x32_bf16(false, a, false, b,
                                                 (short)0, c, false, false);
}

#if HAVE_TDM
// TDM 2-D tile load: tile_d1 rows x tile_d0 elements of a 2-byte-element
// tensor whose innermost dim has length tensor_d0 / stride stride0.
// D# per CDNA5 ISA ch.8: group0 = {count, lds_addr, global_addr, type=2},
// group1 = {data_size=2B, tensor dims, tile dims, dim0 stride}, groups 2/3 = 0.
__device__ __forceinline__ void tdm_load_2d(unsigned lds_addr,
                                            unsigned long long gaddr,
                                            unsigned tile_d0, unsigned tile_d1,
                                            unsigned tensor_d0,
                                            unsigned tensor_d1,
                                            unsigned stride0) {
  v4u g0;
  g0[0] = 1u;                                               // count=1 (user D#)
  g0[1] = lds_addr;                                         // LDS byte address
  g0[2] = (unsigned)gaddr;                                  // global_addr[31:0]
  g0[3] = ((unsigned)(gaddr >> 32) & 0x01FFFFFFu) | 0x80000000u; // [56:32]|type=2
  v8i g1;
  g1[0] = (int)(1u << 16);                                  // data_size=1 (2B)
  g1[1] = (int)(tensor_d0 << 16);                           // tensor_dim0 lo
  g1[2] = (int)((tensor_d0 >> 16) | (tensor_d1 << 16));     // dim0 hi | dim1 lo
  g1[3] = (int)((tensor_d1 >> 16) | (tile_d0 << 16));       // dim1 hi | tile0
  g1[4] = (int)(tile_d1 & 0xFFFFu);                         // tile1 (tile2=0)
  g1[5] = (int)stride0;                                     // dim0 stride lo
  g1[6] = 0;
  g1[7] = 0;
  v4i z4 = {0, 0, 0, 0};
#if defined(__clang_major__) && __clang_major__ >= 23
  v8i z8 = {0, 0, 0, 0, 0, 0, 0, 0};
  __builtin_amdgcn_tensor_load_to_lds(g0, g1, z4, z4, z8, 0);
#else
  __builtin_amdgcn_tensor_load_to_lds(g0, g1, z4, z4, 0);
#endif
}
#endif

// ---------------------------------------------------------------------------
// GroupNorm statistics: one workgroup per (b, g); 16 ch x 4096 = 64K floats.
// ---------------------------------------------------------------------------
__global__ __launch_bounds__(256) void gn_stats_kernel(
    const float* __restrict__ x, float* __restrict__ mean,
    float* __restrict__ rstd) {
  const int bg = blockIdx.x;                 // b*32 + g (channels contiguous)
  const float* xp = x + (size_t)bg * (CPG * N_);
  float s = 0.f, ss = 0.f;
  const int nv4 = CPG * N_ / 4;
  for (int i = threadIdx.x; i < nv4; i += 256) {
    float4 v = ((const float4*)xp)[i];
    s  += v.x + v.y + v.z + v.w;
    ss += v.x * v.x + v.y * v.y + v.z * v.z + v.w * v.w;
  }
  __shared__ float rs[256], rq[256];
  rs[threadIdx.x] = s; rq[threadIdx.x] = ss;
  __syncthreads();
  for (int o = 128; o > 0; o >>= 1) {
    if (threadIdx.x < o) {
      rs[threadIdx.x] += rs[threadIdx.x + o];
      rq[threadIdx.x] += rq[threadIdx.x + o];
    }
    __syncthreads();
  }
  if (threadIdx.x == 0) {
    const float inv = 1.f / (float)(CPG * N_);
    float mu  = rs[0] * inv;
    float var = rq[0] * inv - mu * mu;
    mean[bg] = mu;
    rstd[bg] = rsqrtf(var + EPS_);
  }
}

// ---------------------------------------------------------------------------
// GroupNorm apply: h = (x-mu)*rstd*w + b, stored bf16 [B,C,N].
// ---------------------------------------------------------------------------
__global__ __launch_bounds__(256) void gn_apply_kernel(
    const float* __restrict__ x, const float* __restrict__ gw,
    const float* __restrict__ gb, const float* __restrict__ mean,
    const float* __restrict__ rstd, __bf16* __restrict__ h) {
  const size_t i4 = (size_t)blockIdx.x * 256 + threadIdx.x;  // float4 index
  const size_t e  = i4 * 4;
  const int c  = (int)((e / N_) % C_);
  const int bg = (int)(e / ((size_t)CPG * N_));
  const float mu = mean[bg], rs = rstd[bg];
  const float w = gw[c], b = gb[c];
  float4 v = ((const float4*)x)[i4];
  __bf16* hp = h + e;
  hp[0] = f2bf((v.x - mu) * rs * w + b);
  hp[1] = f2bf((v.y - mu) * rs * w + b);
  hp[2] = f2bf((v.z - mu) * rs * w + b);
  hp[3] = f2bf((v.w - mu) * rs * w + b);
}

// ---------------------------------------------------------------------------
// One-time f32 -> bf16 conversion of the four 512x512 weight matrices.
// ---------------------------------------------------------------------------
__global__ __launch_bounds__(256) void wconv_kernel(
    const float* __restrict__ qw, const float* __restrict__ kw,
    const float* __restrict__ vw, const float* __restrict__ pw,
    __bf16* __restrict__ bq, __bf16* __restrict__ bk,
    __bf16* __restrict__ bv, __bf16* __restrict__ bp) {
  const int i = blockIdx.x * 256 + threadIdx.x;
  bq[i] = f2bf(qw[i]);
  bk[i] = f2bf(kw[i]);
  bv[i] = f2bf(vw[i]);
  bp[i] = f2bf(pw[i]);
}

// ---------------------------------------------------------------------------
// QKV gemm: out[cout, n] = W[cout,:] . h[:, n] + bias.
// Workgroup: 128x128 tile, 8 waves of 32x64. B-tile staged to LDS via the
// Tensor Data Mover (wave 0 issues the DMA, waits TENSORcnt, barrier).
// q and v are stored transposed [N,C] for later contiguous fragment loads.
// ---------------------------------------------------------------------------
__global__ __launch_bounds__(256) void qkv_gemm_kernel(
    const __bf16* __restrict__ h, const __bf16* __restrict__ wq,
    const __bf16* __restrict__ wk, const __bf16* __restrict__ wv,
    const float* __restrict__ qb, const float* __restrict__ kb,
    const float* __restrict__ vb, __bf16* __restrict__ q_t,
    __bf16* __restrict__ k_out, __bf16* __restrict__ v_t) {
  const int tile = blockIdx.x;               // 0..127 : 4 m-tiles x 32 n-tiles
  const int b    = blockIdx.y;
  const int mat  = blockIdx.z;               // 0=q 1=k 2=v
  const int m0 = (tile & 3) * 128;
  const int n0 = (tile >> 2) * 128;
  const __bf16* W    = (mat == 0) ? wq : (mat == 1) ? wk : wv;
  const float*  bias = (mat == 0) ? qb : (mat == 1) ? kb : vb;
  const __bf16* hb = h + (size_t)b * C_ * N_;

  const int wave = threadIdx.x >> 5, lane = threadIdx.x & 31;
  const int wm0 = m0 + (wave >> 1) * 32;
  const int wn0 = n0 + (wave & 1) * 64;

  v8f acc[2][4];
#pragma unroll
  for (int i = 0; i < 2; ++i)
#pragma unroll
    for (int j = 0; j < 4; ++j) acc[i][j] = zero8();

  __shared__ __bf16 bs[32][128];             // 8 KB B-tile
  const int sr = threadIdx.x >> 3, sc = (threadIdx.x & 7) << 4;
  (void)sr; (void)sc;

  for (int ks = 0; ks < C_; ks += 32) {
    __syncthreads();
#if HAVE_TDM
    if (threadIdx.x < 32) {                  // wave 0 drives the TDM
      tdm_load_2d((unsigned)(size_t)(void*)&bs[0][0],
                  (unsigned long long)(size_t)(const void*)
                      &hb[(size_t)ks * N_ + n0],
                  /*tile_d0=*/128u, /*tile_d1=*/32u,
                  /*tensor_d0=*/(unsigned)N_, /*tensor_d1=*/(unsigned)C_,
                  /*stride0=*/(unsigned)N_);
      __builtin_amdgcn_s_wait_tensorcnt(0);
    }
#else
    *(v16bf*)&bs[sr][sc] =
        *(const v16bf*)&hb[(size_t)(ks + sr) * N_ + n0 + sc];
    if (ks + 32 < C_)
      __builtin_prefetch(&hb[(size_t)(ks + 32 + sr) * N_ + n0 + sc], 0, 1);
#endif
    __syncthreads();
#pragma unroll
    for (int rt = 0; rt < 2; ++rt) {
      v16bf a = *(const v16bf*)&W[(size_t)(wm0 + rt * 16 + (lane & 15)) * C_ +
                                  ks + ((lane >> 4) << 4)];
#pragma unroll
      for (int ct = 0; ct < 4; ++ct) {
        v16bf bfr;
        for (int u = 0; u < 16; ++u) bfr[u] = bs[((lane >> 4) << 4) + u][(wn0 - n0) + (ct << 4) + (lane & 15)];
        acc[rt][ct] = wmma_bf16(a, bfr, acc[rt][ct]);
      }
    }
  }

  __bf16* kout = k_out + (size_t)b * C_ * N_;
  __bf16* tout = ((mat == 0) ? q_t : v_t) + (size_t)b * N_ * C_;
#pragma unroll
  for (int rt = 0; rt < 2; ++rt)
#pragma unroll
    for (int ct = 0; ct < 4; ++ct)
#pragma unroll
      for (int e = 0; e < 8; ++e) {
        int row = wm0 + (rt << 4) + e + ((lane >> 4) << 3);
        int col = wn0 + (ct << 4) + (lane & 15);
        float val = acc[rt][ct][e] + bias[row];
        if (mat == 1) kout[(size_t)row * N_ + col] = f2bf(val);
        else          tout[(size_t)col * C_ + row] = f2bf(val);
      }
}

// ---------------------------------------------------------------------------
// Flash pass A: per query row, online max + sumexp over all 4096 keys.
// Workgroup: 128 query rows (8 waves x 16). S = Q.K via WMMA, stats via
// half-wave shuffle reductions (rows 0-7 live in lanes 0-15, 8-15 in 16-31).
// ---------------------------------------------------------------------------
__global__ __launch_bounds__(256) void attn_stats_kernel(
    const __bf16* __restrict__ q_t, const __bf16* __restrict__ k,
    float* __restrict__ mrow, float* __restrict__ lrow) {
  const int b = blockIdx.y;
  const int i0 = blockIdx.x * 128;
  const int wave = threadIdx.x >> 5, lane = threadIdx.x & 31;
  const int iw = i0 + wave * 16;
  const __bf16* qb = q_t + (size_t)b * N_ * C_;
  const __bf16* kb = k + (size_t)b * C_ * N_;

  float mrun[8], lrun[8];
#pragma unroll
  for (int e = 0; e < 8; ++e) { mrun[e] = -1e30f; lrun[e] = 0.f; }

  __shared__ __bf16 kst[32][128];
  const int sr = threadIdx.x >> 3, sc = (threadIdx.x & 7) << 4;

  for (int j0 = 0; j0 < N_; j0 += 128) {
    v8f s[8];
#pragma unroll
    for (int i = 0; i < 8; ++i) s[i] = zero8();
    for (int kc = 0; kc < C_; kc += 32) {
      __syncthreads();
      *(v16bf*)&kst[sr][sc] =
          *(const v16bf*)&kb[(size_t)(kc + sr) * N_ + j0 + sc];
      __syncthreads();
      v16bf a = *(const v16bf*)&qb[(size_t)(iw + (lane & 15)) * C_ + kc +
                                   ((lane >> 4) << 4)];
#pragma unroll
      for (int ct = 0; ct < 8; ++ct) {
        v16bf bfr;
        for (int u = 0; u < 16; ++u) bfr[u] = kst[((lane >> 4) << 4) + u][(ct << 4) + (lane & 15)];
        s[ct] = wmma_bf16(a, bfr, s[ct]);
      }
    }
#pragma unroll
    for (int e = 0; e < 8; ++e) {
      float mb = -1e30f;
#pragma unroll
      for (int ct = 0; ct < 8; ++ct) mb = fmaxf(mb, s[ct][e] * SCALE_);
#pragma unroll
      for (int off = 8; off >= 1; off >>= 1)
        mb = fmaxf(mb, __shfl_xor(mb, off, 16));
      float sm = 0.f;
#pragma unroll
      for (int ct = 0; ct < 8; ++ct) sm += __expf(s[ct][e] * SCALE_ - mb);
#pragma unroll
      for (int off = 8; off >= 1; off >>= 1) sm += __shfl_xor(sm, off, 16);
      float mn = fmaxf(mrun[e], mb);
      lrun[e] = lrun[e] * __expf(mrun[e] - mn) + sm * __expf(mb - mn);
      mrun[e] = mn;
    }
  }
  if ((lane & 15) == 0) {
#pragma unroll
    for (int e = 0; e < 8; ++e) {
      int row = iw + e + ((lane >> 4) << 3);
      mrow[b * N_ + row] = mrun[e];
      lrow[b * N_ + row] = lrun[e];
    }
  }
}

// ---------------------------------------------------------------------------
// Flash pass B: recompute S, P = exp(S*scale - m)/l via per-wave LDS tile,
// O += P.V with V staged in LDS. O kept in registers (16x256 per wave).
// grid.z splits the 512 output channels into 2 blocks of 256.
// ---------------------------------------------------------------------------
__global__ __launch_bounds__(256) void attn_out_kernel(
    const __bf16* __restrict__ q_t, const __bf16* __restrict__ k,
    const __bf16* __restrict__ v_t, const float* __restrict__ mrow,
    const float* __restrict__ lrow, __bf16* __restrict__ o_t) {
  const int b = blockIdx.y;
  const int i0 = blockIdx.x * 128;
  const int c0 = blockIdx.z * 256;
  const int wave = threadIdx.x >> 5, lane = threadIdx.x & 31;
  const int iw = i0 + wave * 16;
  const __bf16* qb = q_t + (size_t)b * N_ * C_;
  const __bf16* kb = k + (size_t)b * C_ * N_;
  const __bf16* vb = v_t + (size_t)b * N_ * C_;

  float mv[8], lv[8];
#pragma unroll
  for (int e = 0; e < 8; ++e) {
    int row = iw + e + ((lane >> 4) << 3);
    mv[e] = mrow[b * N_ + row];
    lv[e] = 1.f / lrow[b * N_ + row];
  }

  v8f o[16];
#pragma unroll
  for (int i = 0; i < 16; ++i) o[i] = zero8();

  __shared__ __bf16 stage[8192];             // 16 KB: k-tile / v-tile (union)
  __shared__ __bf16 pls[8][16][128];         // 32 KB: per-wave P tiles
  const int sr = threadIdx.x >> 3, sc = (threadIdx.x & 7) << 4;
  const int vr = threadIdx.x & 127, vh = (threadIdx.x >> 7) << 5;

  for (int j0 = 0; j0 < N_; j0 += 128) {
    // ---- S = Q.K for this 128-wide key block --------------------------------
    v8f s[8];
#pragma unroll
    for (int i = 0; i < 8; ++i) s[i] = zero8();
    for (int kc = 0; kc < C_; kc += 32) {
      __syncthreads();
      *(v16bf*)&stage[sr * 128 + sc] =
          *(const v16bf*)&kb[(size_t)(kc + sr) * N_ + j0 + sc];
      if (kc + 32 < C_)
        __builtin_prefetch(&kb[(size_t)(kc + 32 + sr) * N_ + j0 + sc], 0, 1);
      __syncthreads();
      v16bf a = *(const v16bf*)&qb[(size_t)(iw + (lane & 15)) * C_ + kc +
                                   ((lane >> 4) << 4)];
#pragma unroll
      for (int ct = 0; ct < 8; ++ct) {
        v16bf bfr;
        for (int u = 0; u < 16; ++u) bfr[u] = stage[(size_t)(((lane >> 4) << 4) + u) * 128 + (ct << 4) + (lane & 15)];
        s[ct] = wmma_bf16(a, bfr, s[ct]);
      }
    }
    // ---- P = exp(S*scale - m) / l, to per-wave LDS (same-wave ds in-order) --
#pragma unroll
    for (int e = 0; e < 8; ++e) {
      int pr = e + ((lane >> 4) << 3);
#pragma unroll
      for (int ct = 0; ct < 8; ++ct) {
        float p = __expf(s[ct][e] * SCALE_ - mv[e]) * lv[e];
        pls[wave][pr][(ct << 4) + (lane & 15)] = f2bf(p);
      }
    }
    // ---- O += P.V : 4 chunks of 64 channels ---------------------------------
    for (int cc = 0; cc < 4; ++cc) {
      __syncthreads();
      *(v16bf*)&stage[vr * 64 + vh] =
          *(const v16bf*)&vb[(size_t)(j0 + vr) * C_ + c0 + (cc << 6) + vh];
      *(v16bf*)&stage[vr * 64 + vh + 16] =
          *(const v16bf*)&vb[(size_t)(j0 + vr) * C_ + c0 + (cc << 6) + vh + 16];
      __syncthreads();
#pragma unroll
      for (int kj = 0; kj < 4; ++kj) {
        v16bf a = *(const v16bf*)&pls[wave][lane & 15]
                                    [(kj << 5) + ((lane >> 4) << 4)];
#pragma unroll
        for (int ct = 0; ct < 4; ++ct) {
          int oi = (cc << 2) + ct;
          v16bf bfr;
          for (int u = 0; u < 16; ++u) bfr[u] = stage[(size_t)((kj << 5) + ((lane >> 4) << 4) + u) * 64 + (ct << 4) + (lane & 15)];
          o[oi] = wmma_bf16(a, bfr, o[oi]);
        }
      }
    }
  }

  __bf16* ot = o_t + (size_t)b * N_ * C_;
#pragma unroll
  for (int t = 0; t < 16; ++t)
#pragma unroll
    for (int e = 0; e < 8; ++e) {
      int row = iw + e + ((lane >> 4) << 3);
      int col = c0 + (t << 4) + (lane & 15);
      ot[(size_t)row * C_ + col] = f2bf(o[t][e]);
    }
}

// ---------------------------------------------------------------------------
// Projection + residual: out[b,d,n] = x + proj_w[d,:].o[:,n] + proj_b[d].
// B-tile transposed from o_t [N,C] through LDS.
// ---------------------------------------------------------------------------
__global__ __launch_bounds__(256) void proj_kernel(
    const __bf16* __restrict__ o_t, const __bf16* __restrict__ wp,
    const float* __restrict__ pb, const float* __restrict__ x,
    float* __restrict__ out) {
  const int tile = blockIdx.x;
  const int b    = blockIdx.y;
  const int m0 = (tile & 3) * 128;
  const int n0 = (tile >> 2) * 128;
  const int wave = threadIdx.x >> 5, lane = threadIdx.x & 31;
  const int wm0 = m0 + (wave >> 1) * 32;
  const int wn0 = n0 + (wave & 1) * 64;
  const __bf16* ob = o_t + (size_t)b * N_ * C_;

  v8f acc[2][4];
#pragma unroll
  for (int i = 0; i < 2; ++i)
#pragma unroll
    for (int j = 0; j < 4; ++j) acc[i][j] = zero8();

  __shared__ __bf16 bs[32][128];
  const int tj = threadIdx.x >> 1, tp = (threadIdx.x & 1) << 4;

  for (int kc = 0; kc < C_; kc += 32) {
    __syncthreads();
    {
      v16bf val = *(const v16bf*)&ob[(size_t)(n0 + tj) * C_ + kc + tp];
#pragma unroll
      for (int e = 0; e < 16; ++e) bs[tp + e][tj] = val[e];
    }
    __syncthreads();
#pragma unroll
    for (int rt = 0; rt < 2; ++rt) {
      v16bf a = *(const v16bf*)&wp[(size_t)(wm0 + rt * 16 + (lane & 15)) * C_ +
                                   kc + ((lane >> 4) << 4)];
#pragma unroll
      for (int ct = 0; ct < 4; ++ct) {
        v16bf bfr;
        for (int u = 0; u < 16; ++u) bfr[u] = bs[((lane >> 4) << 4) + u][(wn0 - n0) + (ct << 4) + (lane & 15)];
        acc[rt][ct] = wmma_bf16(a, bfr, acc[rt][ct]);
      }
    }
  }

  const float* xb = x + (size_t)b * C_ * N_;
  float* yb = out + (size_t)b * C_ * N_;
#pragma unroll
  for (int rt = 0; rt < 2; ++rt)
#pragma unroll
    for (int ct = 0; ct < 4; ++ct)
#pragma unroll
      for (int e = 0; e < 8; ++e) {
        int row = wm0 + (rt << 4) + e + ((lane >> 4) << 3);
        int col = wn0 + (ct << 4) + (lane & 15);
        size_t idx = (size_t)row * N_ + col;
        yb[idx] = xb[idx] + acc[rt][ct][e] + pb[row];
      }
}

// ---------------------------------------------------------------------------
extern "C" void kernel_launch(void* const* d_in, const int* in_sizes, int n_in,
                              void* d_out, int out_size, void* d_ws,
                              size_t ws_size, hipStream_t stream) {
  (void)in_sizes; (void)n_in; (void)out_size; (void)ws_size;
  const float* x   = (const float*)d_in[0];
  const float* gnw = (const float*)d_in[1];
  const float* gnb = (const float*)d_in[2];
  const float* qw  = (const float*)d_in[3];
  const float* qb  = (const float*)d_in[4];
  const float* kw  = (const float*)d_in[5];
  const float* kbv = (const float*)d_in[6];
  const float* vw  = (const float*)d_in[7];
  const float* vbv = (const float*)d_in[8];
  const float* pw  = (const float*)d_in[9];
  const float* pbv = (const float*)d_in[10];
  float* out = (float*)d_out;

  char* ws = (char*)d_ws;
  const size_t SZH = (size_t)B_ * C_ * N_ * 2;           // 32 MB bf16 tensor
  __bf16* h   = (__bf16*)(ws);
  __bf16* q_t = (__bf16*)(ws + SZH);
  __bf16* kk  = (__bf16*)(ws + 2 * SZH);
  __bf16* v_t = (__bf16*)(ws + 3 * SZH);
  __bf16* o_t = (__bf16*)(ws + 4 * SZH);
  __bf16* wq  = (__bf16*)(ws + 5 * SZH);
  __bf16* wk  = wq + (size_t)C_ * C_;
  __bf16* wv  = wk + (size_t)C_ * C_;
  __bf16* wp  = wv + (size_t)C_ * C_;
  float* mean = (float*)(ws + 5 * SZH + 4 * (size_t)C_ * C_ * 2);
  float* rstd = mean + B_ * G_;
  float* mrow = rstd + B_ * G_;
  float* lrow = mrow + (size_t)B_ * N_;

  gn_stats_kernel<<<dim3(B_ * G_), dim3(256), 0, stream>>>(x, mean, rstd);
  gn_apply_kernel<<<dim3((B_ * C_ * N_ / 4) / 256), dim3(256), 0, stream>>>(
      x, gnw, gnb, mean, rstd, h);
  wconv_kernel<<<dim3(C_ * C_ / 256), dim3(256), 0, stream>>>(
      qw, kw, vw, pw, wq, wk, wv, wp);
  qkv_gemm_kernel<<<dim3(128, B_, 3), dim3(256), 0, stream>>>(
      h, wq, wk, wv, qb, kbv, vbv, q_t, kk, v_t);
  attn_stats_kernel<<<dim3(N_ / 128, B_), dim3(256), 0, stream>>>(
      q_t, kk, mrow, lrow);
  attn_out_kernel<<<dim3(N_ / 128, B_, 2), dim3(256), 0, stream>>>(
      q_t, kk, v_t, mrow, lrow, o_t);
  proj_kernel<<<dim3(128, B_), dim3(256), 0, stream>>>(o_t, wp, pbv, x, out);
}
